// AttentionBlock_90933047591481
// MI455X (gfx1250) — matrix operand order repaired and checked by reading, not compiled, not run on hardware
//
#include <hip/hip_runtime.h>

// ---------------------------------------------------------------- constants
static constexpr int Hd  = 1024;   // hidden
static constexpr int Sq  = 2048;   // seq len
static constexpr int Bb  = 2;      // batch
static constexpr int NH  = 16;     // heads
static constexpr int HD  = 64;     // head dim
static constexpr int FF  = 4096;   // ffn dim
static constexpr int MR  = Bb * Sq;  // 4096 rows

typedef __bf16 bf16_t;
typedef __attribute__((ext_vector_type(16))) __bf16 v16bf;
typedef __attribute__((ext_vector_type(8)))  __bf16 v8bf;
typedef __attribute__((ext_vector_type(8)))  float  v8f;

union BF16x16 { v16bf v; v8bf h[2]; };

__device__ __forceinline__ v8f wmma_bf16(const BF16x16& a, const BF16x16& b, v8f c) {
  return __builtin_amdgcn_wmma_f32_16x16x32_bf16(false, a.v, false, b.v,
                                                 (short)0, c, false, false);
}

__device__ __forceinline__ v8bf load8(const bf16_t* base, unsigned elemOff) {
  return *(const v8bf*)(base + elemOff);   // uniform base + 32-bit offset
}

// ------------------------------------------------ weight convert+transpose
// W [K x N] f32 row-major  ->  Wt [N x K] bf16 row-major
__global__ void wt_convert_kernel(const float* __restrict__ W,
                                  bf16_t* __restrict__ Wt, int K, int N) {
  unsigned idx = blockIdx.x * blockDim.x + threadIdx.x;
  unsigned total = (unsigned)K * (unsigned)N;
  if (idx >= total) return;
  unsigned n = idx / K;
  unsigned k = idx % K;
  Wt[idx] = (bf16_t)W[k * (unsigned)N + n];
}

// v [B*S, H] bf16 -> vt [(b*NH+h)*HD + d, S] bf16
__global__ void vtrans_kernel(const bf16_t* __restrict__ v,
                              bf16_t* __restrict__ vt) {
  unsigned idx = blockIdx.x * blockDim.x + threadIdx.x;
  if (idx >= (unsigned)MR * Hd) return;
  unsigned s   = idx % Sq;
  unsigned row = idx / Sq;      // (b*NH+h)*HD + d
  unsigned d   = row % HD;
  unsigned bh  = row / HD;
  unsigned h   = bh % NH;
  unsigned b   = bh / NH;
  vt[idx] = v[(b * Sq + s) * Hd + h * HD + d];
}

// ------------------------------------------------------------- layernorm
__global__ void ln_bf16_kernel(const float* __restrict__ x,
                               const float* __restrict__ g,
                               const float* __restrict__ bta,
                               bf16_t* __restrict__ out) {
  __shared__ float red[256];
  const int tid = threadIdx.x;
  const unsigned row = blockIdx.x;
  const float* xr = x + (size_t)row * Hd;
  float s = 0.f, ss = 0.f;
  for (int i = tid; i < Hd; i += 256) { float v = xr[i]; s += v; ss += v * v; }
  red[tid] = s; __syncthreads();
  for (int off = 128; off > 0; off >>= 1) {
    if (tid < off) red[tid] += red[tid + off];
    __syncthreads();
  }
  float mean = red[0] * (1.0f / Hd);
  __syncthreads();
  red[tid] = ss; __syncthreads();
  for (int off = 128; off > 0; off >>= 1) {
    if (tid < off) red[tid] += red[tid + off];
    __syncthreads();
  }
  float var  = red[0] * (1.0f / Hd) - mean * mean;
  float rstd = rsqrtf(var + 1e-5f);
  bf16_t* orow = out + (size_t)row * Hd;
  for (int i = tid; i < Hd; i += 256)
    orow[i] = (bf16_t)((xr[i] - mean) * rstd * g[i] + bta[i]);
}

// ------------------------------------------------------------- WMMA GEMM
// C[M,N] = A[M,K](bf16,row) * Bt[N,K](bf16,row == B col-major) + bias (+res)
// 4 waves/block; each wave owns a 32x64 strip (2 M-tiles x 4 N-tiles).
// k-loop manually unrolled 2x with two named operand register sets so loads
// for k+32 overlap the WMMAs for k. 32-bit offsets + launch_bounds keep the
// allocator from spilling operands.
template <bool OUT_BF16, bool RELU, bool RES>
__global__ __launch_bounds__(128, 1)
void gemm_wmma_kernel(const bf16_t* __restrict__ A,
                      const bf16_t* __restrict__ Bt,
                      const float*  __restrict__ bias,
                      const float*  __restrict__ res,
                      void* __restrict__ Cp, int M, int N, int K) {
  const int lane = threadIdx.x & 31;
  const int wave = threadIdx.x >> 5;
  const int l15  = lane & 15;
  const int half = lane >> 4;
  const int tileM = blockIdx.y * 128 + wave * 32;
  const int tileN = blockIdx.x * 64;

  v8f acc[2][4];
#pragma unroll
  for (int m = 0; m < 2; ++m)
#pragma unroll
    for (int t = 0; t < 4; ++t)
#pragma unroll
      for (int j = 0; j < 8; ++j) acc[m][t][j] = 0.f;

  const unsigned aoff0 = (unsigned)(tileM + l15) * K;
  const unsigned aoff1 = (unsigned)(tileM + 16 + l15) * K;
  const unsigned boffBase = (unsigned)(tileN + l15) * K + half * 8;

  BF16x16 a0[2], b0[4], a1[2], b1[4];

  auto load_ops = [&](BF16x16 (&aa)[2], BF16x16 (&bb)[4], int k) {
    aa[0].h[0] = load8(A, aoff0 + k + half * 8);       // K = half*8 + 0..7
    aa[0].h[1] = load8(A, aoff0 + k + 16 + half * 8);  // K = 16 + half*8 + 0..7
    aa[1].h[0] = load8(A, aoff1 + k + half * 8);
    aa[1].h[1] = load8(A, aoff1 + k + 16 + half * 8);
#pragma unroll
    for (int t = 0; t < 4; ++t) {
      unsigned boff = boffBase + (unsigned)(t * 16) * K + k;
      bb[t].h[0] = load8(Bt, boff);                    // K = half*8 + 0..7
      bb[t].h[1] = load8(Bt, boff + 16);               // K = 16 + half*8 + 0..7
    }
  };
  auto do_mma = [&](BF16x16 (&aa)[2], BF16x16 (&bb)[4]) {
#pragma unroll
    for (int t = 0; t < 4; ++t) {
      acc[0][t] = wmma_bf16(aa[0], bb[t], acc[0][t]);
      acc[1][t] = wmma_bf16(aa[1], bb[t], acc[1][t]);
    }
  };

  load_ops(a0, b0, 0);
  for (int k = 0; k < K; k += 64) {   // K is a multiple of 64
    load_ops(a1, b1, k + 32);
    do_mma(a0, b0);
    if (k + 64 < K) load_ops(a0, b0, k + 64);
    do_mma(a1, b1);
  }

#pragma unroll
  for (int m = 0; m < 2; ++m)
#pragma unroll
    for (int t = 0; t < 4; ++t) {
      const int col = tileN + t * 16 + l15;
      const float bv = bias[col];
#pragma unroll
      for (int j = 0; j < 8; ++j) {
        const unsigned row = tileM + m * 16 + half * 8 + j;
        float v = acc[m][t][j] + bv;
        if (RELU) v = fmaxf(v, 0.f);
        if (RES)  v += res[row * (unsigned)N + col];
        if (OUT_BF16) ((bf16_t*)Cp)[row * (unsigned)N + col] = (bf16_t)v;
        else          ((float*)Cp)[row * (unsigned)N + col] = v;
      }
    }
}

// ------------------------------------------------------- flash attention
// grid: (Sq/16, Bb*NH), block: 32 (one wave). 16 queries per wave,
// streams keys in tiles of 32 with online softmax.
// K operands are double-buffered (next tile's K prefetched during softmax).
// V operands use a single short-lived buffer whose loads issue right after
// the score WMMAs, so the softmax LDS/VALU section hides their latency.
__global__ __launch_bounds__(32, 1)
void attn_kernel(const bf16_t* __restrict__ q,
                 const bf16_t* __restrict__ kmat,
                 const bf16_t* __restrict__ vt,
                 bf16_t* __restrict__ ctx) {
  __shared__ float sP[16 * 32];
  __shared__ float sC[16];
  __shared__ float sL[16];

  const int lane = threadIdx.x;
  const int l15  = lane & 15;
  const int half = lane >> 4;
  const int bh = blockIdx.y;
  const int b  = bh / NH;
  const int h  = bh % NH;
  const int q0 = blockIdx.x * 16;
  const float scale = 0.125f;   // 1/sqrt(64)

  // Q as two A-operands (d 0..31 and 32..63), loaded once
  const unsigned qoff = (unsigned)(b * Sq + q0 + l15) * Hd + h * HD;
  BF16x16 qa0, qa1;
  qa0.h[0] = load8(q, qoff + half * 8);
  qa0.h[1] = load8(q, qoff + 16 + half * 8);
  qa1.h[0] = load8(q, qoff + 32 + half * 8);
  qa1.h[1] = load8(q, qoff + 48 + half * 8);

  v8f o[4];
#pragma unroll
  for (int t = 0; t < 4; ++t)
#pragma unroll
    for (int j = 0; j < 8; ++j) o[t][j] = 0.f;

  float mrow = -3.0e38f, lrow = 0.f;  // valid in lanes 0..15 (row = lane)

  // Per-lane constant offset parts (32-bit): varying per-tile part is kt-based.
  const unsigned kRowOff = (unsigned)(b * Sq + l15) * Hd + h * HD + half * 8;
  const unsigned vRowOff = (unsigned)(bh * HD + l15) * Sq + half * 8;

  // K operands: kk[0],kk[1] = sub-tile 0 (d 0..31 / 32..63);
  //             kk[2],kk[3] = sub-tile 1.
  auto loadK = [&](BF16x16 (&kk)[4], int kt) {
#pragma unroll
    for (int st = 0; st < 2; ++st) {
      unsigned off = kRowOff + (unsigned)(kt + st * 16) * Hd;
      kk[2 * st + 0].h[0] = load8(kmat, off);          // d = half*8 + 0..7
      kk[2 * st + 0].h[1] = load8(kmat, off + 16);     // d = 16 + half*8 + 0..7
      kk[2 * st + 1].h[0] = load8(kmat, off + 32);     // d = 32 + half*8 + 0..7
      kk[2 * st + 1].h[1] = load8(kmat, off + 48);     // d = 48 + half*8 + 0..7
    }
  };

  BF16x16 kA[4], kB[4];

  auto tile_step = [&](BF16x16 (&kk)[4], BF16x16 (&kn)[4], int kt, bool pref) {
    // ---- scores S = Q K^T for 32 keys (two 16-col tiles)
    v8f s0, s1;
#pragma unroll
    for (int j = 0; j < 8; ++j) { s0[j] = 0.f; s1[j] = 0.f; }
    s0 = wmma_bf16(qa0, kk[0], s0);
    s0 = wmma_bf16(qa1, kk[1], s0);
    s1 = wmma_bf16(qa0, kk[2], s1);
    s1 = wmma_bf16(qa1, kk[3], s1);

    // ---- V operands for THIS tile: independent of softmax, issued now so
    // the softmax section below hides their latency (short live range).
    BF16x16 vv[4];
#pragma unroll
    for (int nt = 0; nt < 4; ++nt) {
      unsigned off = vRowOff + (unsigned)(nt * 16) * Sq + kt;
      vv[nt].h[0] = load8(vt, off);
      vv[nt].h[1] = load8(vt, off + 16);
    }

    // ---- prefetch next tile's K operands (overlaps softmax below)
    if (pref) loadK(kn, kt + 32);

#pragma unroll
    for (int j = 0; j < 8; ++j) {
      sP[(half * 8 + j) * 32 + 0  + l15] = s0[j] * scale;
      sP[(half * 8 + j) * 32 + 16 + l15] = s1[j] * scale;
    }
    __syncthreads();

    // ---- online softmax stats (row r owned by lane r, r<16)
    if (lane < 16) {
      float* srow = &sP[lane * 32];
      float mx = mrow;
      for (int i = 0; i < 32; ++i) mx = fmaxf(mx, srow[i]);
      float c = __expf(mrow - mx);
      float sum = 0.f;
      for (int i = 0; i < 32; ++i) {
        float pe = __expf(srow[i] - mx);
        srow[i] = pe;
        sum += pe;
      }
      lrow = lrow * c + sum;
      mrow = mx;
      sC[lane] = c;
    }
    __syncthreads();

    // ---- rescale accumulators
    float c8[8];
#pragma unroll
    for (int j = 0; j < 8; ++j) c8[j] = sC[half * 8 + j];
#pragma unroll
    for (int t = 0; t < 4; ++t)
#pragma unroll
      for (int j = 0; j < 8; ++j) o[t][j] *= c8[j];

    // ---- P (16x32) -> bf16 A-operand layout
    BF16x16 pa;
#pragma unroll
    for (int e = 0; e < 8; ++e) {
      pa.v[e]     = (bf16_t)sP[l15 * 32 + half * 8 + e];
      pa.v[8 + e] = (bf16_t)sP[l15 * 32 + 16 + half * 8 + e];
    }

    // ---- O += P * V_tile (V loads have had the whole softmax to land)
#pragma unroll
    for (int nt = 0; nt < 4; ++nt)
      o[nt] = wmma_bf16(pa, vv[nt], o[nt]);
    __syncthreads();
  };

  loadK(kA, 0);
  for (int kt = 0; kt < Sq; kt += 64) {      // Sq is a multiple of 64
    tile_step(kA, kB, kt, true);                       // prefetches kt+32
    tile_step(kB, kA, kt + 32, kt + 64 < Sq);          // prefetches kt+64
  }

  if (lane < 16) sL[lane] = 1.f / lrow;
  __syncthreads();
  float inv[8];
#pragma unroll
  for (int j = 0; j < 8; ++j) inv[j] = sL[half * 8 + j];
#pragma unroll
  for (int nt = 0; nt < 4; ++nt)
#pragma unroll
    for (int j = 0; j < 8; ++j)
      ctx[(unsigned)(b * Sq + q0 + half * 8 + j) * Hd + h * HD + nt * 16 + l15] =
          (bf16_t)(o[nt][j] * inv[j]);
}

// ---------------------------------------------------------------- driver
extern "C" void kernel_launch(void* const* d_in, const int* in_sizes, int n_in,
                              void* d_out, int out_size, void* d_ws, size_t ws_size,
                              hipStream_t stream) {
  (void)in_sizes; (void)n_in; (void)out_size; (void)ws_size;
  const float* x   = (const float*)d_in[0];
  const float* Wq  = (const float*)d_in[1];
  const float* bq  = (const float*)d_in[2];
  const float* Wk  = (const float*)d_in[3];
  const float* bk  = (const float*)d_in[4];
  const float* Wv  = (const float*)d_in[5];
  const float* bv  = (const float*)d_in[6];
  const float* Wo  = (const float*)d_in[7];
  const float* bo  = (const float*)d_in[8];
  const float* g1  = (const float*)d_in[9];
  const float* b1  = (const float*)d_in[10];
  const float* W1  = (const float*)d_in[11];
  const float* bf1 = (const float*)d_in[12];
  const float* W2  = (const float*)d_in[13];
  const float* bf2 = (const float*)d_in[14];
  const float* g2  = (const float*)d_in[15];
  const float* b2  = (const float*)d_in[16];
  float* out = (float*)d_out;

  // ---- workspace bump allocator
  char* p = (char*)d_ws;
  auto alloc = [&](size_t bytes) -> void* {
    void* r = (void*)p;
    p += (bytes + 255) & ~(size_t)255;
    return r;
  };
  bf16_t* xn1  = (bf16_t*)alloc((size_t)MR * Hd * 2);
  bf16_t* xn2  = (bf16_t*)alloc((size_t)MR * Hd * 2);
  bf16_t* WqT  = (bf16_t*)alloc((size_t)Hd * Hd * 2);
  bf16_t* WkT  = (bf16_t*)alloc((size_t)Hd * Hd * 2);
  bf16_t* WvT  = (bf16_t*)alloc((size_t)Hd * Hd * 2);
  bf16_t* WoT  = (bf16_t*)alloc((size_t)Hd * Hd * 2);
  bf16_t* W1T  = (bf16_t*)alloc((size_t)Hd * FF * 2);
  bf16_t* W2T  = (bf16_t*)alloc((size_t)FF * Hd * 2);
  bf16_t* qb   = (bf16_t*)alloc((size_t)MR * Hd * 2);
  bf16_t* kb   = (bf16_t*)alloc((size_t)MR * Hd * 2);
  bf16_t* vb   = (bf16_t*)alloc((size_t)MR * Hd * 2);
  bf16_t* vtb  = (bf16_t*)alloc((size_t)MR * Hd * 2);
  bf16_t* ctx  = (bf16_t*)alloc((size_t)MR * Hd * 2);
  float*  out1 = (float*) alloc((size_t)MR * Hd * 4);
  bf16_t* ffn1 = (bf16_t*)alloc((size_t)MR * FF * 2);

  const int cvtBlk = 256;
  const int nHH = (Hd * Hd + cvtBlk - 1) / cvtBlk;
  const int nHF = (Hd * FF + cvtBlk - 1) / cvtBlk;

  // ---- weight convert + transpose to bf16 [N x K]
  wt_convert_kernel<<<nHH, cvtBlk, 0, stream>>>(Wq, WqT, Hd, Hd);
  wt_convert_kernel<<<nHH, cvtBlk, 0, stream>>>(Wk, WkT, Hd, Hd);
  wt_convert_kernel<<<nHH, cvtBlk, 0, stream>>>(Wv, WvT, Hd, Hd);
  wt_convert_kernel<<<nHH, cvtBlk, 0, stream>>>(Wo, WoT, Hd, Hd);
  wt_convert_kernel<<<nHF, cvtBlk, 0, stream>>>(W1, W1T, Hd, FF);  // -> [FF x Hd]
  wt_convert_kernel<<<nHF, cvtBlk, 0, stream>>>(W2, W2T, FF, Hd);  // -> [Hd x FF]

  // ---- LN1
  ln_bf16_kernel<<<MR, 256, 0, stream>>>(x, g1, b1, xn1);

  // ---- QKV projections (bf16 out + bias)
  dim3 gHH(Hd / 64, MR / 128);
  gemm_wmma_kernel<true, false, false><<<gHH, 128, 0, stream>>>(
      xn1, WqT, bq, nullptr, qb, MR, Hd, Hd);
  gemm_wmma_kernel<true, false, false><<<gHH, 128, 0, stream>>>(
      xn1, WkT, bk, nullptr, kb, MR, Hd, Hd);
  gemm_wmma_kernel<true, false, false><<<gHH, 128, 0, stream>>>(
      xn1, WvT, bv, nullptr, vb, MR, Hd, Hd);

  // ---- transpose V per head for PV WMMA
  vtrans_kernel<<<(MR * Hd + cvtBlk - 1) / cvtBlk, cvtBlk, 0, stream>>>(vb, vtb);

  // ---- attention
  dim3 gAtt(Sq / 16, Bb * NH);
  attn_kernel<<<gAtt, 32, 0, stream>>>(qb, kb, vtb, ctx);

  // ---- output projection + residual (f32)
  gemm_wmma_kernel<false, false, true><<<gHH, 128, 0, stream>>>(
      ctx, WoT, bo, x, out1, MR, Hd, Hd);

  // ---- LN2
  ln_bf16_kernel<<<MR, 256, 0, stream>>>(out1, g2, b2, xn2);

  // ---- FFN: relu(xn2 @ W1 + bf1) -> bf16
  dim3 gHF(FF / 64, MR / 128);
  gemm_wmma_kernel<true, true, false><<<gHF, 128, 0, stream>>>(
      xn2, W1T, bf1, nullptr, ffn1, MR, FF, Hd);

  // ---- FFN out + residual -> d_out (f32)
  gemm_wmma_kernel<false, false, true><<<gHH, 128, 0, stream>>>(
      ffn1, W2T, bf2, out1, out, MR, Hd, FF);
}
